// OffsetModel_14920716386528
// MI455X (gfx1250) — matrix operand order repaired and checked by reading, not compiled, not run on hardware
//
#include <hip/hip_runtime.h>
#include <hip/hip_bf16.h>

typedef __attribute__((ext_vector_type(16))) __bf16 v16bf;
typedef __attribute__((ext_vector_type(8)))  float  v8f;

#define CEILDIV(a, b) (((a) + (b) - 1) / (b))

__device__ inline unsigned short f32_to_bf16_bits(float f) {
  union { float f; unsigned u; } a; a.f = f;
  unsigned u = a.u;
  u += 0x7fffu + ((u >> 16) & 1u);   // round-to-nearest-even
  return (unsigned short)(u >> 16);
}
__device__ inline __bf16 bits_to_bf16(unsigned short s) {
  union { unsigned short s; __bf16 b; } c; c.s = s; return c.b;
}

// ---------------------------------------------------------------------------
// Weight prep: f32 [K][N] row-major  ->  bf16 [Np][Kp] (N-major, zero padded)
// ---------------------------------------------------------------------------
__global__ void k_prep_w(const float* __restrict__ W, unsigned short* __restrict__ Wt,
                         int K, int N, int Kp, int Np) {
  int i = blockIdx.x * blockDim.x + threadIdx.x;
  int total = Kp * Np;
  if (i >= total) return;
  int n = i / Kp, k = i - n * Kp;
  float v = (k < K && n < N) ? W[(size_t)k * N + n] : 0.0f;
  Wt[(size_t)n * Kp + k] = f32_to_bf16_bits(v);
}

// ---------------------------------------------------------------------------
// Strided point extraction: dst[i] = src[i*stride]
// ---------------------------------------------------------------------------
__global__ void k_stride_pts(const float* __restrict__ src, float* __restrict__ dst,
                             int n, int stride) {
  int i = blockIdx.x * blockDim.x + threadIdx.x;
  if (i >= n * 3) return;
  int p = i / 3, c = i - p * 3;
  dst[i] = src[(size_t)p * stride * 3 + c];
}

// ---------------------------------------------------------------------------
// KNN, K=16 (indices only). One thread per query, streaming top-16.
// ---------------------------------------------------------------------------
__global__ void k_knn16(const float* __restrict__ q, const float* __restrict__ t,
                        int nq, int nt, int* __restrict__ idx) {
  int i = blockIdx.x * blockDim.x + threadIdx.x;
  if (i >= nq) return;
  float qx = q[i * 3], qy = q[i * 3 + 1], qz = q[i * 3 + 2];
  float bd[16]; int bi[16];
#pragma unroll
  for (int j = 0; j < 16; ++j) { bd[j] = 3.4e38f; bi[j] = 0; }
  for (int s = 0; s < nt; ++s) {
    int sp = s + 96; if (sp >= nt) sp = nt - 1;
    __builtin_prefetch(&t[(size_t)sp * 3], 0, 0);   // global_prefetch_b8
    float dx = t[s * 3] - qx, dy = t[s * 3 + 1] - qy, dz = t[s * 3 + 2] - qz;
    float d2 = dx * dx + dy * dy + dz * dz;
    float worst = bd[0]; int wj = 0;
#pragma unroll
    for (int j = 1; j < 16; ++j) if (bd[j] > worst) { worst = bd[j]; wj = j; }
    if (d2 < worst) {
#pragma unroll
      for (int j = 0; j < 16; ++j) if (j == wj) { bd[j] = d2; bi[j] = s; }
    }
  }
#pragma unroll
  for (int j = 0; j < 16; ++j) idx[(size_t)i * 16 + j] = bi[j];
}

// ---------------------------------------------------------------------------
// KNN, K=3, with squared distances (for inverse-distance upsampling)
// ---------------------------------------------------------------------------
__global__ void k_knn3(const float* __restrict__ q, const float* __restrict__ t,
                       int nq, int nt, int* __restrict__ idx, float* __restrict__ d2o) {
  int i = blockIdx.x * blockDim.x + threadIdx.x;
  if (i >= nq) return;
  float qx = q[i * 3], qy = q[i * 3 + 1], qz = q[i * 3 + 2];
  float bd[3]; int bi[3];
#pragma unroll
  for (int j = 0; j < 3; ++j) { bd[j] = 3.4e38f; bi[j] = 0; }
  for (int s = 0; s < nt; ++s) {
    float dx = t[s * 3] - qx, dy = t[s * 3 + 1] - qy, dz = t[s * 3 + 2] - qz;
    float d2 = dx * dx + dy * dy + dz * dz;
    float worst = bd[0]; int wj = 0;
#pragma unroll
    for (int j = 1; j < 3; ++j) if (bd[j] > worst) { worst = bd[j]; wj = j; }
    if (d2 < worst) {
#pragma unroll
      for (int j = 0; j < 3; ++j) if (j == wj) { bd[j] = d2; bi[j] = s; }
    }
  }
#pragma unroll
  for (int j = 0; j < 3; ++j) { idx[(size_t)i * 3 + j] = bi[j]; d2o[(size_t)i * 3 + j] = bd[j]; }
}

// ---------------------------------------------------------------------------
// Inverse-distance weighted 3-NN interpolation, C channels
// ---------------------------------------------------------------------------
__global__ void k_interp(const int* __restrict__ idx, const float* __restrict__ d2,
                         const float* __restrict__ tf, float* __restrict__ qf,
                         int nq, int C) {
  int i = blockIdx.x * blockDim.x + threadIdx.x;
  if (i >= nq * C) return;
  int qi = i / C, c = i - qi * C;
  float w0 = 1.0f / (d2[(size_t)qi * 3 + 0] + 1e-8f);
  float w1 = 1.0f / (d2[(size_t)qi * 3 + 1] + 1e-8f);
  float w2 = 1.0f / (d2[(size_t)qi * 3 + 2] + 1e-8f);
  float ws = w0 + w1 + w2;
  float v = w0 * tf[(size_t)idx[(size_t)qi * 3 + 0] * C + c]
          + w1 * tf[(size_t)idx[(size_t)qi * 3 + 1] * C + c]
          + w2 * tf[(size_t)idx[(size_t)qi * 3 + 2] * C + c];
  qf[(size_t)i] = v / ws;
}

// ---------------------------------------------------------------------------
// Fragment loaders (contiguous k = 16*half + e order, same for A and B)
// ---------------------------------------------------------------------------
__device__ inline v16bf load_a_frag(const unsigned short* Arow /* row m base */,
                                    int Kp, int kt, int half) {
  v16bf a;
#pragma unroll
  for (int e = 0; e < 16; ++e) {
    int kl = half * 16 + e;
    a[e] = bits_to_bf16(Arow[kt * 32 + kl]);
  }
  return a;
}
__device__ inline v16bf load_b_frag(const unsigned short* __restrict__ Wt,
                                    int n, int Kp, int kt, int half) {
  const unsigned short* wp = &Wt[(size_t)n * Kp + kt * 32 + half * 16];
  v16bf b;
#pragma unroll
  for (int e = 0; e < 16; ++e) b[e] = bits_to_bf16(wp[e]);
  return b;
}

// ---------------------------------------------------------------------------
// gs_conv: one wave per query point, 4 N-tiles register-blocked per pass.
// Cout/16 is always a multiple of 4 here (128/192/256) -> no guards needed.
//   A (16 x Kp)  = 16 neighbor rows of concat([pre]relu(src_feats), rel_pos)
//   out[q, n]    = max over 16 neighbor rows of relu(A @ Wn + b)  (+= if accum)
// ---------------------------------------------------------------------------
#define GS_KPMAX 288
__global__ void __launch_bounds__(128) k_gsconv(
    const float* __restrict__ q_pts, const float* __restrict__ s_pts,
    const float* __restrict__ s_f, const int* __restrict__ idx,
    const unsigned short* __restrict__ Wt, const float* __restrict__ bias,
    float* __restrict__ out,
    int nq, int Cin, int Cout, int Kp, int preact, int accum) {
  __shared__ unsigned short lds[4 * 16 * GS_KPMAX];
  int lane = threadIdx.x & 31;
  int wave = threadIdx.x >> 5;
  int q = blockIdx.x * 4 + wave;
  unsigned short* Aw = &lds[wave * 16 * GS_KPMAX];

  if (q < nq) {
    float qx = q_pts[(size_t)q * 3], qy = q_pts[(size_t)q * 3 + 1], qz = q_pts[(size_t)q * 3 + 2];
    for (int e = lane; e < 16 * Kp; e += 32) {
      int r = e / Kp, k = e - r * Kp;
      int s = idx[(size_t)q * 16 + r];
      float v = 0.0f;
      if (k < Cin) {
        v = s_f[(size_t)s * Cin + k];
        if (preact) v = fmaxf(v, 0.0f);
      } else if (k < Cin + 3) {
        int c = k - Cin;
        float qp = (c == 0) ? qx : ((c == 1) ? qy : qz);
        v = s_pts[(size_t)s * 3 + c] - qp;
      }
      Aw[r * Kp + k] = f32_to_bf16_bits(v);
    }
  }
  __syncthreads();
  if (q >= nq) return;   // wave-uniform: EXEC stays all-ones for WMMA

  int m = lane & 15, half = lane >> 4;
  int ntiles = Cout >> 4, ktiles = Kp >> 5;
  const unsigned short* Arow = &Aw[m * Kp];
  const v8f vzero = {};

  for (int ntb = 0; ntb < ntiles; ntb += 4) {
    v8f acc[4];
#pragma unroll
    for (int j = 0; j < 4; ++j) acc[j] = vzero;
    for (int kt = 0; kt < ktiles; ++kt) {
      v16bf a = load_a_frag(Arow, Kp, kt, half);     // reused for 4 N tiles
#pragma unroll
      for (int j = 0; j < 4; ++j) {
        v16bf b = load_b_frag(Wt, (ntb + j) * 16 + m, Kp, kt, half);
        acc[j] = __builtin_amdgcn_wmma_f32_16x16x32_bf16(
            false, a, false, b, (short)0, acc[j], false, false);
      }
    }
#pragma unroll
    for (int j = 0; j < 4; ++j) {
      int n = (ntb + j) * 16 + m;
      float bv = bias[n];
      float mx = 0.0f;                 // relu'd values are >= 0
#pragma unroll
      for (int r = 0; r < 8; ++r) mx = fmaxf(mx, fmaxf(acc[j][r] + bv, 0.0f));
      mx = fmaxf(mx, __shfl_xor(mx, 16, 32));        // combine row halves
      if (half == 0) {
        size_t o = (size_t)q * Cout + n;
        if (accum) out[o] += mx; else out[o] = mx;
      }
    }
  }
}

// ---------------------------------------------------------------------------
// Generic GEMM: out[M,Nreal] (=|+=) act(A[M,K] @ W + bias)
//   block = 4 waves; block handles one 16-row M tile;
//   wave handles JN consecutive N tiles (compile-time, fully unrolled:
//   one A fragment feeds JN independent WMMAs, no guards, no phi copies).
// ---------------------------------------------------------------------------
#define GEMM_KPMAX 288
template <int JN>
__global__ void __launch_bounds__(128) k_gemm(
    const float* __restrict__ Ag, const unsigned short* __restrict__ Wt,
    const float* __restrict__ bias, float* __restrict__ out,
    int M, int K, int Kp, int Nreal, int Ntiles,
    int relu, int accum, int has_bias) {
  __shared__ unsigned short As[16 * GEMM_KPMAX];
  int tid = threadIdx.x;
  int lane = tid & 31, wave = tid >> 5;
  int mt = blockIdx.x;
  for (int e = tid; e < 16 * Kp; e += 128) {
    int r = e / Kp, k = e - r * Kp;
    int row = mt * 16 + r;
    float v = (row < M && k < K) ? Ag[(size_t)row * K + k] : 0.0f;
    As[r * Kp + k] = f32_to_bf16_bits(v);
  }
  __syncthreads();
  int ntb = (blockIdx.y * 4 + wave) * JN;  // first of this wave's JN N tiles
  if (ntb >= Ntiles) return;               // wave-uniform; Ntiles % JN == 0

  int m = lane & 15, half = lane >> 4;
  int ktiles = Kp >> 5;
  const unsigned short* Arow = &As[m * Kp];
  const v8f vzero = {};
  v8f acc[JN];
#pragma unroll
  for (int j = 0; j < JN; ++j) acc[j] = vzero;
  for (int kt = 0; kt < ktiles; ++kt) {
    v16bf a = load_a_frag(Arow, Kp, kt, half);
#pragma unroll
    for (int j = 0; j < JN; ++j) {
      v16bf b = load_b_frag(Wt, (ntb + j) * 16 + m, Kp, kt, half);
      acc[j] = __builtin_amdgcn_wmma_f32_16x16x32_bf16(
          false, a, false, b, (short)0, acc[j], false, false);
    }
  }
#pragma unroll
  for (int j = 0; j < JN; ++j) {
    int n = (ntb + j) * 16 + m;
    float bv = (has_bias && n < Nreal) ? bias[n] : 0.0f;
#pragma unroll
    for (int r = 0; r < 8; ++r) {
      int row = mt * 16 + r + half * 8;
      if (n < Nreal && row < M) {
        float v = acc[j][r] + bv;
        if (relu) v = fmaxf(v, 0.0f);
        size_t o = (size_t)row * Nreal + n;
        if (accum) out[o] += v; else out[o] = v;
      }
    }
  }
}

// ---------------------------------------------------------------------------
__global__ void k_add(float* __restrict__ out, const float* __restrict__ a, int n) {
  int i = blockIdx.x * blockDim.x + threadIdx.x;
  if (i < n) out[i] += a[i];
}

// ===========================================================================
extern "C" void kernel_launch(void* const* d_in, const int* in_sizes, int n_in,
                              void* d_out, int out_size, void* d_ws, size_t ws_size,
                              hipStream_t stream) {
  (void)in_sizes; (void)n_in; (void)out_size; (void)ws_size;
  const float* points   = (const float*)d_in[0];
  const float* features = (const float*)d_in[1];
  const float *enc_Wn[4], *enc_b[4], *up_Wn[4], *up_b[4], *up_Wq[4];
  const float *lin_W[3], *lin_b[3], *mlp_W[4], *mlp_b[4];
  for (int i = 0; i < 4; ++i) { enc_Wn[i] = (const float*)d_in[2 + i];  enc_b[i] = (const float*)d_in[6 + i]; }
  for (int i = 0; i < 4; ++i) { up_Wn[i]  = (const float*)d_in[10 + i]; up_b[i]  = (const float*)d_in[14 + i];
                                up_Wq[i]  = (const float*)d_in[18 + i]; }
  for (int i = 0; i < 3; ++i) { lin_W[i]  = (const float*)d_in[22 + i]; lin_b[i] = (const float*)d_in[25 + i]; }
  for (int i = 0; i < 4; ++i) { mlp_W[i]  = (const float*)d_in[28 + i]; mlp_b[i] = (const float*)d_in[32 + i]; }

  const int n0 = 16384, n1 = 4096, n2 = 1024, n3 = 256, n4 = 64;

  size_t off = 0; char* base = (char*)d_ws;
  auto alloc = [&](size_t bytes) -> void* {
    void* p = base + off; off += (bytes + 255) & ~(size_t)255; return p;
  };

  // bf16 transposed/padded weights
  unsigned short* wEnc[4]; unsigned short* wUpN[4]; unsigned short* wUpQ[4];
  unsigned short* wLin[3]; unsigned short* wMlp[4];
  const int encK[4]  = { 9, 131, 195, 259 }, encKp[4] = { 32, 160, 224, 288 };
  const int encN[4]  = { 128, 192, 256, 256 };
  const int linK[3]  = { 6, 128, 192 },      linKp[3] = { 32, 128, 192 };
  const int mlpK[4]  = { 256, 128, 64, 32 }, mlpN[4]  = { 128, 64, 32, 3 };
  const int mlpNp[4] = { 128, 64, 32, 16 };
  for (int l = 0; l < 4; ++l) wEnc[l] = (unsigned short*)alloc((size_t)encKp[l] * encN[l] * 2);
  for (int l = 0; l < 4; ++l) wUpN[l] = (unsigned short*)alloc((size_t)288 * 256 * 2);
  for (int l = 0; l < 4; ++l) wUpQ[l] = (unsigned short*)alloc((size_t)256 * 256 * 2);
  for (int l = 0; l < 3; ++l) wLin[l] = (unsigned short*)alloc((size_t)linKp[l] * 256 * 2);
  for (int l = 0; l < 4; ++l) wMlp[l] = (unsigned short*)alloc((size_t)mlpK[l] * mlpNp[l] * 2);

  float* p1 = (float*)alloc((size_t)n1 * 3 * 4);
  float* p2 = (float*)alloc((size_t)n2 * 3 * 4);
  float* p3 = (float*)alloc((size_t)n3 * 3 * 4);
  float* p4 = (float*)alloc((size_t)n4 * 3 * 4);
  float* f1 = (float*)alloc((size_t)n1 * 128 * 4);
  float* f2 = (float*)alloc((size_t)n2 * 192 * 4);
  float* f3 = (float*)alloc((size_t)n3 * 256 * 4);
  float* f4 = (float*)alloc((size_t)n4 * 256 * 4);
  float* fe3 = (float*)alloc((size_t)n3 * 256 * 4);
  float* fe2 = (float*)alloc((size_t)n2 * 256 * 4);
  float* fe1 = (float*)alloc((size_t)n1 * 256 * 4);
  float* fe0 = (float*)alloc((size_t)n0 * 256 * 4);
  float* qf  = (float*)alloc((size_t)n0 * 256 * 4);  // reused as MLP h1
  float* h2  = (float*)alloc((size_t)n0 * 64 * 4);
  float* h3  = (float*)alloc((size_t)n0 * 32 * 4);
  int*   idx16 = (int*)alloc((size_t)n0 * 16 * 4);
  int*   idx3  = (int*)alloc((size_t)n0 * 3 * 4);
  float* d3    = (float*)alloc((size_t)n0 * 3 * 4);

  auto prep = [&](const float* W, unsigned short* Wt, int K, int N, int Kp, int Np) {
    int total = Kp * Np;
    k_prep_w<<<CEILDIV(total, 256), 256, 0, stream>>>(W, Wt, K, N, Kp, Np);
  };
  for (int l = 0; l < 4; ++l) prep(enc_Wn[l], wEnc[l], encK[l], encN[l], encKp[l], encN[l]);
  for (int l = 0; l < 4; ++l) prep(up_Wn[l],  wUpN[l], 259, 256, 288, 256);
  for (int l = 0; l < 4; ++l) prep(up_Wq[l],  wUpQ[l], 256, 256, 256, 256);
  for (int l = 0; l < 3; ++l) prep(lin_W[l],  wLin[l], linK[l], 256, linKp[l], 256);
  for (int l = 0; l < 4; ++l) prep(mlp_W[l],  wMlp[l], mlpK[l], mlpN[l], mlpK[l], mlpNp[l]);

  auto gsconv = [&](const float* qp, const float* sp, const float* sf, const int* idx,
                    const unsigned short* Wt, const float* b, float* o,
                    int nq, int Cin, int Cout, int Kp, int preact, int accum) {
    k_gsconv<<<nq / 4, 128, 0, stream>>>(qp, sp, sf, idx, Wt, b, o, nq, Cin, Cout, Kp, preact, accum);
  };
  auto gemm = [&](const float* A, const unsigned short* Wt, const float* b, float* o,
                  int M, int K, int Kp, int Nreal, int Np, int relu, int accum) {
    int Ntiles = Np / 16;
    int hb = b ? 1 : 0;
    if (Ntiles % 4 == 0) {
      dim3 grid(M / 16, CEILDIV(Ntiles, 16));
      k_gemm<4><<<grid, 128, 0, stream>>>(A, Wt, b, o, M, K, Kp, Nreal, Ntiles, relu, accum, hb);
    } else if (Ntiles % 2 == 0) {
      dim3 grid(M / 16, CEILDIV(Ntiles, 8));
      k_gemm<2><<<grid, 128, 0, stream>>>(A, Wt, b, o, M, K, Kp, Nreal, Ntiles, relu, accum, hb);
    } else {
      dim3 grid(M / 16, CEILDIV(Ntiles, 4));
      k_gemm<1><<<grid, 128, 0, stream>>>(A, Wt, b, o, M, K, Kp, Nreal, Ntiles, relu, accum, hb);
    }
  };
  auto knn16 = [&](const float* q, const float* t, int nq, int nt) {
    k_knn16<<<CEILDIV(nq, 256), 256, 0, stream>>>(q, t, nq, nt, idx16);
  };
  auto knn3 = [&](const float* q, const float* t, int nq, int nt) {
    k_knn3<<<CEILDIV(nq, 256), 256, 0, stream>>>(q, t, nq, nt, idx3, d3);
  };
  auto interp = [&](const float* tf, float* o, int nq) {
    k_interp<<<CEILDIV(nq * 256, 256), 256, 0, stream>>>(idx3, d3, tf, o, nq, 256);
  };

  // ---- strided point levels ----
  k_stride_pts<<<CEILDIV(n1 * 3, 256), 256, 0, stream>>>(points, p1, n1, 4);
  k_stride_pts<<<CEILDIV(n2 * 3, 256), 256, 0, stream>>>(points, p2, n2, 16);
  k_stride_pts<<<CEILDIV(n3 * 3, 256), 256, 0, stream>>>(points, p3, n3, 64);
  k_stride_pts<<<CEILDIV(n4 * 3, 256), 256, 0, stream>>>(points, p4, n4, 256);

  // ---- encoder ----
  knn16(p1, points, n1, n0);
  gsconv(p1, points, features, idx16, wEnc[0], enc_b[0], f1, n1, 6,   128, 32,  0, 0);
  knn16(p2, p1, n2, n1);
  gsconv(p2, p1, f1, idx16, wEnc[1], enc_b[1], f2, n2, 128, 192, 160, 1, 0);
  knn16(p3, p2, n3, n2);
  gsconv(p3, p2, f2, idx16, wEnc[2], enc_b[2], f3, n3, 192, 256, 224, 1, 0);
  knn16(p4, p3, n4, n3);
  gsconv(p4, p3, f3, idx16, wEnc[3], enc_b[3], f4, n4, 256, 256, 288, 1, 0);

  // ---- decoder layer 3 (q = p3, src = p4/f4) ----
  knn3(p3, p4, n3, n4);
  interp(f4, qf, n3);
  gemm(qf, wUpQ[3], nullptr, fe3, n3, 256, 256, 256, 256, 0, 0);
  knn16(p3, p4, n3, n4);
  gsconv(p3, p4, f4, idx16, wUpN[3], up_b[3], fe3, n3, 256, 256, 288, 1, 1);
  k_add<<<CEILDIV(n3 * 256, 256), 256, 0, stream>>>(fe3, f3, n3 * 256);

  // ---- decoder layer 2 (q = p2, src = p3/fe3) ----
  knn3(p2, p3, n2, n3);
  interp(fe3, qf, n2);
  gemm(qf, wUpQ[2], nullptr, fe2, n2, 256, 256, 256, 256, 0, 0);
  knn16(p2, p3, n2, n3);
  gsconv(p2, p3, fe3, idx16, wUpN[2], up_b[2], fe2, n2, 256, 256, 288, 1, 1);
  gemm(f2, wLin[2], lin_b[2], fe2, n2, 192, 192, 256, 256, 0, 1);

  // ---- decoder layer 1 (q = p1, src = p2/fe2) ----
  knn3(p1, p2, n1, n2);
  interp(fe2, qf, n1);
  gemm(qf, wUpQ[1], nullptr, fe1, n1, 256, 256, 256, 256, 0, 0);
  knn16(p1, p2, n1, n2);
  gsconv(p1, p2, fe2, idx16, wUpN[1], up_b[1], fe1, n1, 256, 256, 288, 1, 1);
  gemm(f1, wLin[1], lin_b[1], fe1, n1, 128, 128, 256, 256, 0, 1);

  // ---- decoder layer 0 (q = points, src = p1/fe1) ----
  knn3(points, p1, n0, n1);
  interp(fe1, qf, n0);
  gemm(qf, wUpQ[0], nullptr, fe0, n0, 256, 256, 256, 256, 0, 0);
  knn16(points, p1, n0, n1);
  gsconv(points, p1, fe1, idx16, wUpN[0], up_b[0], fe0, n0, 256, 256, 288, 1, 1);
  gemm(features, wLin[0], lin_b[0], fe0, n0, 6, 32, 256, 256, 0, 1);

  // ---- MLP head ----
  float* h1 = qf;                         // qf is free now
  gemm(fe0, wMlp[0], mlp_b[0], h1, n0, 256, 256, 128, 128, 1, 0);
  gemm(h1,  wMlp[1], mlp_b[1], h2, n0, 128, 128, 64,  64,  1, 0);
  gemm(h2,  wMlp[2], mlp_b[2], h3, n0, 64,  64,  32,  32,  1, 0);
  gemm(h3,  wMlp[3], mlp_b[3], (float*)d_out, n0, 32, 32, 3, 16, 0, 0);
}
